// GATModel_40123584479614
// MI455X (gfx1250) — matrix operand order repaired and checked by reading, not compiled, not run on hardware
//
#include <hip/hip_runtime.h>
#include <hip/hip_bf16.h>

#define BN_EPS 1e-5f
#define NEG_SLOPE 0.2f

typedef __attribute__((ext_vector_type(16))) _Float16 v16h;
typedef __attribute__((ext_vector_type(8)))  float    v8f;

// ---------------------------------------------------------------------------
// WMMA GEMM: C[M,Nc] = A[M,K] @ B[K,Nc]  (f32 in/out, f16 multiply, f32 acc)
// Block = NW waves, block tile = 32 rows x (NW*16) cols.
// Each wave owns one 16-col strip and two 16x16 accumulators (rows 0-15,16-31),
// reusing its B fragment across both WMMAs per K-step.
// LDS tiles are stored pre-swizzled into WMMA fragment order (ISA 7.12.2), so
// each fragment is one contiguous 32B per-lane read (2x ds_load_b128).
//   A frag (lane l, elem i): k = (i<8 ? 8*hi + i : 16 + 8*hi + (i-8)), m=l&15,
//                            hi=l>>4  -> inverse: lane=m+16*((k>>3)&1), i=(k&7)+8*(k>=16)
//   B frag (lane l, elem i): k = (i<8 ? 8*hi + i : 16 + 8*hi + (i-8)), n = l&15
//                            -> inverse: lane=(n&15)+16*((k>>3)&1), i=(k&7)+8*(k>=16)
// K % 32 == 0, Nc % (NW*16) == 0 required.
// ---------------------------------------------------------------------------
template <int NW>
__global__ void gemm_wmma_f16(const float* __restrict__ A,
                              const float* __restrict__ B,
                              float* __restrict__ Cmat,
                              int M, int K, int Nc,
                              const float* __restrict__ bias, int do_relu) {
    __shared__ __align__(32) _Float16 As[2][32 * 16];  // [m-subtile][lane*16+i]
    __shared__ __align__(32) _Float16 Bs[4][32 * 16];  // [wave][lane*16+i]

    const int tid  = threadIdx.x;
    const int lane = tid & 31;
    const int wave = tid >> 5;
    const int nth  = 32 * NW;
    const int row0 = blockIdx.y * 32;
    const int col0 = blockIdx.x * (NW * 16);

    v8f acc0 = {}, acc1 = {};

    for (int kt = 0; kt < K; kt += 32) {
        // ---- stage A: 32 rows x 32 k, float4 along k, swizzled f16 store ----
        for (int idx = tid; idx < 32 * 8; idx += nth) {
            int r   = idx >> 3;           // row in tile 0..31
            int c4  = (idx & 7) << 2;     // k offset 0,4,...,28
            int grow = row0 + r;
            float4 v = make_float4(0.f, 0.f, 0.f, 0.f);
            if (grow < M) {
                const float* ap = A + (size_t)grow * K + kt + c4;
                v = *(const float4*)ap;
                if (kt + 32 < K)  // prefetch next K-tile (global_prefetch_b8)
                    __builtin_prefetch(ap + 32, 0, 3);
            }
            int sub = r >> 4, m = r & 15;
            int hi  = (c4 >> 3) & 1;                     // lane-hi group
            int i0  = ((c4 & 16) ? 8 : 0) + (c4 & 7);    // fragment elem base
            _Float16* dst = &As[sub][(m + 16 * hi) * 16 + i0];
            dst[0] = (_Float16)v.x; dst[1] = (_Float16)v.y;
            dst[2] = (_Float16)v.z; dst[3] = (_Float16)v.w;
        }
        // ---- stage B: 32 k x NW*16 n, float4 along n, swizzled f16 store ----
        for (int idx = tid; idx < 32 * NW * 4; idx += nth) {
            int k  = idx / (NW * 4);
            int n4 = (idx % (NW * 4)) << 2;
            float4 v = *(const float4*)(B + (size_t)(kt + k) * Nc + col0 + n4);
            int wv    = n4 >> 4;
            int lane0 = (n4 & 15) + 16 * ((k >> 3) & 1);
            int i     = (k & 7) + ((k & 16) ? 8 : 0);
            Bs[wv][(lane0 + 0) * 16 + i] = (_Float16)v.x;
            Bs[wv][(lane0 + 1) * 16 + i] = (_Float16)v.y;
            Bs[wv][(lane0 + 2) * 16 + i] = (_Float16)v.z;
            Bs[wv][(lane0 + 3) * 16 + i] = (_Float16)v.w;
        }
        __syncthreads();

        // contiguous 32B fragment loads (2x ds_load_b128 each)
        v16h bf = *(const v16h*)&Bs[wave][lane * 16];
        v16h a0 = *(const v16h*)&As[0][lane * 16];
        v16h a1 = *(const v16h*)&As[1][lane * 16];

        acc0 = __builtin_amdgcn_wmma_f32_16x16x32_f16(
            false, a0, false, bf, (short)0, acc0, false, false);
        acc1 = __builtin_amdgcn_wmma_f32_16x16x32_f16(
            false, a1, false, bf, (short)0, acc1, false, false);
        __syncthreads();
    }

    // ---- store D: VGPR v -> row v + 8*(lane>=16), col = lane&15 ----
    const int ncol = col0 + wave * 16 + (lane & 15);
    const float badd = bias ? bias[ncol] : 0.0f;
#pragma unroll
    for (int v = 0; v < 8; ++v) {
        int r0 = row0 + v + 8 * (lane >> 4);
        if (r0 < M) {
            float val = acc0[v] + badd;
            if (do_relu) val = val > 0.0f ? val : 0.0f;
            Cmat[(size_t)r0 * Nc + ncol] = val;
        }
        int r1 = r0 + 16;
        if (r1 < M) {
            float val = acc1[v] + badd;
            if (do_relu) val = val > 0.0f ? val : 0.0f;
            Cmat[(size_t)r1 * Nc + ncol] = val;
        }
    }
}

// ---------------------------------------------------------------------------
// Utility: fill float buffer
// ---------------------------------------------------------------------------
__global__ void fill_kernel(float* __restrict__ p, long n, float v) {
    long t = (long)blockIdx.x * blockDim.x + threadIdx.x;
    if (t < n) p[t] = v;
}

// ---------------------------------------------------------------------------
// Per-node attention scores: s_src[n,h] = sum_c h[n,h*C+c]*a_s[h,c]; same dst
// ---------------------------------------------------------------------------
__global__ void scores_kernel(const float* __restrict__ h,
                              const float* __restrict__ a_s,
                              const float* __restrict__ a_d,
                              float* __restrict__ ssrc, float* __restrict__ sdst,
                              int Nn, int H, int C) {
    int t = blockIdx.x * blockDim.x + threadIdx.x;
    if (t >= Nn * H) return;
    int n = t / H, hh = t % H;
    int ldh = H * C;
    const float* hp = h + (size_t)n * ldh + hh * C;
    float s0 = 0.0f, s1 = 0.0f;
    for (int c = 0; c < C; ++c) {
        float v = hp[c];
        s0 += v * a_s[hh * C + c];
        s1 += v * a_d[hh * C + c];
    }
    ssrc[t] = s0;
    sdst[t] = s1;
}

__device__ __forceinline__ float lrelu(float x) {
    return x > 0.0f ? x : NEG_SLOPE * x;
}

__device__ __forceinline__ void atomicMaxF(float* addr, float v) {
    if (v >= 0.0f) atomicMax((int*)addr, __float_as_int(v));
    else           atomicMin((unsigned int*)addr, __float_as_uint(v));
}

// ---------------------------------------------------------------------------
// Pass 1: segment max over incoming edges (incl. implicit self loops)
// ---------------------------------------------------------------------------
__global__ void edge_max_kernel(const int* __restrict__ ei, int E, int Etot,
                                const float* __restrict__ ssrc,
                                const float* __restrict__ sdst,
                                float* __restrict__ mbuf, int H) {
    int t = blockIdx.x * blockDim.x + threadIdx.x;
    if (t >= Etot) return;
    int s, d;
    if (t < E) { s = ei[t]; d = ei[E + t]; } else { s = t - E; d = s; }
    for (int hh = 0; hh < H; ++hh) {
        float e = lrelu(ssrc[s * H + hh] + sdst[d * H + hh]);
        atomicMaxF(&mbuf[d * H + hh], e);
    }
}

// ---------------------------------------------------------------------------
// Pass 2: w = exp(e - max); denom[dst] += w; agg[dst] += w * h[src]
// ---------------------------------------------------------------------------
__global__ void edge_acc_kernel(const int* __restrict__ ei, int E, int Etot,
                                const float* __restrict__ ssrc,
                                const float* __restrict__ sdst,
                                const float* __restrict__ mbuf,
                                float* __restrict__ denom,
                                const float* __restrict__ hfeat,
                                float* __restrict__ agg, int H, int C) {
    long t = (long)blockIdx.x * blockDim.x + threadIdx.x;
    long tot = (long)Etot * H;
    if (t >= tot) return;
    int e  = (int)(t / H);
    int hh = (int)(t % H);
    int s, d;
    if (e < E) { s = ei[e]; d = ei[E + e]; } else { s = e - E; d = s; }
    float ev = lrelu(ssrc[s * H + hh] + sdst[d * H + hh]);
    float w  = __expf(ev - mbuf[d * H + hh]);
    atomicAdd(&denom[d * H + hh], w);
    int ldh = H * C;
    const float* hs = hfeat + (size_t)s * ldh + hh * C;
    float* ad = agg + (size_t)d * ldh + hh * C;
    for (int c = 0; c < C; ++c) atomicAdd(&ad[c], w * hs[c]);
}

// ---------------------------------------------------------------------------
// Finalize: agg /= denom, + bias  (concat layout; layer3 has H=1 so identical)
// ---------------------------------------------------------------------------
__global__ void finalize_kernel(float* __restrict__ agg,
                                const float* __restrict__ denom,
                                const float* __restrict__ bias,
                                int Nn, int H, int C) {
    int outdim = H * C;
    long t = (long)blockIdx.x * blockDim.x + threadIdx.x;
    if (t >= (long)Nn * outdim) return;
    int n = (int)(t / outdim), j = (int)(t % outdim);
    int hh = j / C;
    agg[t] = agg[t] / denom[n * H + hh] + bias[j];
}

// ---------------------------------------------------------------------------
// BatchNorm stats: one block per channel, population mean/var
// ---------------------------------------------------------------------------
__global__ void bn_stats_kernel(const float* __restrict__ x, int Nn, int ld,
                                float* __restrict__ mu, float* __restrict__ var) {
    __shared__ float sh[256], shq[256];
    int j = blockIdx.x;
    float s = 0.0f, sq = 0.0f;
    for (int r = threadIdx.x; r < Nn; r += blockDim.x) {
        float v = x[(size_t)r * ld + j];
        s += v; sq += v * v;
    }
    sh[threadIdx.x] = s; shq[threadIdx.x] = sq;
    __syncthreads();
    for (int off = 128; off > 0; off >>= 1) {
        if ((int)threadIdx.x < off) {
            sh[threadIdx.x]  += sh[threadIdx.x + off];
            shq[threadIdx.x] += shq[threadIdx.x + off];
        }
        __syncthreads();
    }
    if (threadIdx.x == 0) {
        float m = sh[0] / (float)Nn;
        mu[j]  = m;
        var[j] = shq[0] / (float)Nn - m * m;
    }
}

// BN apply + ReLU, in place
__global__ void bn_apply_kernel(float* __restrict__ x,
                                const float* __restrict__ g,
                                const float* __restrict__ be,
                                const float* __restrict__ mu,
                                const float* __restrict__ var,
                                int Nn, int ld) {
    long t = (long)blockIdx.x * blockDim.x + threadIdx.x;
    if (t >= (long)Nn * ld) return;
    int j = (int)(t % ld);
    float y = g[j] * (x[t] - mu[j]) * rsqrtf(var[j] + BN_EPS) + be[j];
    x[t] = y > 0.0f ? y : 0.0f;
}

// Final tiny head: out[n,:2] = h[n,:32] @ hw2 + hb2
__global__ void mlp2_kernel(const float* __restrict__ hin,
                            const float* __restrict__ w2,
                            const float* __restrict__ b2,
                            float* __restrict__ out, int Nn) {
    int n = blockIdx.x * blockDim.x + threadIdx.x;
    if (n >= Nn) return;
    float a0 = b2[0], a1 = b2[1];
    const float* hp = hin + (size_t)n * 32;
    for (int c = 0; c < 32; ++c) {
        float v = hp[c];
        a0 += v * w2[c * 2 + 0];
        a1 += v * w2[c * 2 + 1];
    }
    out[n * 2 + 0] = a0;
    out[n * 2 + 1] = a1;
}

// ---------------------------------------------------------------------------
extern "C" void kernel_launch(void* const* d_in, const int* in_sizes, int n_in,
                              void* d_out, int out_size, void* d_ws, size_t ws_size,
                              hipStream_t stream) {
    (void)n_in; (void)out_size; (void)ws_size;

    const float* x   = (const float*)d_in[0];
    const int*   ei  = (const int*)d_in[1];
    const float* hw1 = (const float*)d_in[26];
    const float* hb1 = (const float*)d_in[27];
    const float* hw2 = (const float*)d_in[28];
    const float* hb2 = (const float*)d_in[29];

    const int Nn   = in_sizes[0] / 128;   // 50000
    const int E    = in_sizes[1] / 2;     // 400000
    const int Etot = E + Nn;              // with self-loops

    // workspace layout (floats)
    float* ws       = (float*)d_ws;
    float* hbuf     = ws;                                   // N x 256
    float* aggbuf   = hbuf   + (size_t)Nn * 256;            // N x 256
    float* ssrc     = aggbuf + (size_t)Nn * 256;            // N x 4
    float* sdst     = ssrc   + (size_t)Nn * 4;
    float* mbuf     = sdst   + (size_t)Nn * 4;
    float* denom    = mbuf   + (size_t)Nn * 4;
    float* stat_mu  = denom  + (size_t)Nn * 4;              // 256
    float* stat_var = stat_mu + 256;                        // 256

    const int incs[4] = {128, 256, 256, 256};
    const int Hs[4]   = {4, 4, 4, 1};
    const int C       = 64;

    const int TPB = 256;
    const float* curx = x;

    for (int l = 0; l < 4; ++l) {
        const float* W   = (const float*)d_in[2 + 6 * l];
        const float* a_s = (const float*)d_in[3 + 6 * l];
        const float* a_d = (const float*)d_in[4 + 6 * l];
        const float* bb  = (const float*)d_in[5 + 6 * l];
        const float* gg  = (const float*)d_in[6 + 6 * l];
        const float* bee = (const float*)d_in[7 + 6 * l];

        const int H = Hs[l], inc = incs[l], outdim = H * C;

        // 1) projection h = curx @ W  (WMMA, 32x64 block tile, 4 waves)
        {
            dim3 grid(outdim / 64, (Nn + 31) / 32);
            gemm_wmma_f16<4><<<grid, 128, 0, stream>>>(curx, W, hbuf, Nn, inc,
                                                       outdim, nullptr, 0);
        }
        // 2) per-node scores
        scores_kernel<<<(Nn * H + TPB - 1) / TPB, TPB, 0, stream>>>(
            hbuf, a_s, a_d, ssrc, sdst, Nn, H, C);
        // 3) init segment buffers
        fill_kernel<<<(Nn * H + TPB - 1) / TPB, TPB, 0, stream>>>(mbuf,
            (long)Nn * H, -3.0e38f);
        fill_kernel<<<(Nn * H + TPB - 1) / TPB, TPB, 0, stream>>>(denom,
            (long)Nn * H, 0.0f);
        {
            long tot = (long)Nn * outdim;
            fill_kernel<<<(unsigned)((tot + TPB - 1) / TPB), TPB, 0, stream>>>(
                aggbuf, tot, 0.0f);
        }
        // 4) segment softmax + weighted aggregation
        edge_max_kernel<<<(Etot + TPB - 1) / TPB, TPB, 0, stream>>>(
            ei, E, Etot, ssrc, sdst, mbuf, H);
        {
            long tot = (long)Etot * H;
            edge_acc_kernel<<<(unsigned)((tot + TPB - 1) / TPB), TPB, 0, stream>>>(
                ei, E, Etot, ssrc, sdst, mbuf, denom, hbuf, aggbuf, H, C);
        }
        // 5) divide + bias
        {
            long tot = (long)Nn * outdim;
            finalize_kernel<<<(unsigned)((tot + TPB - 1) / TPB), TPB, 0, stream>>>(
                aggbuf, denom, bb, Nn, H, C);
        }
        // 6) BatchNorm + ReLU in place
        bn_stats_kernel<<<outdim, 256, 0, stream>>>(aggbuf, Nn, outdim, stat_mu,
                                                    stat_var);
        {
            long tot = (long)Nn * outdim;
            bn_apply_kernel<<<(unsigned)((tot + TPB - 1) / TPB), TPB, 0, stream>>>(
                aggbuf, gg, bee, stat_mu, stat_var, Nn, outdim);
        }
        curx = aggbuf;
    }

    // MLP head: relu(h @ hw1 + hb1) -> [N,32] (WMMA), then @ hw2 + hb2 -> [N,2]
    {
        dim3 grid(1, (Nn + 31) / 32);
        gemm_wmma_f16<2><<<grid, 64, 0, stream>>>(aggbuf, hw1, hbuf, Nn, 64, 32,
                                                  hb1, 1);
    }
    mlp2_kernel<<<(Nn + TPB - 1) / TPB, TPB, 0, stream>>>(hbuf, hw2, hb2,
                                                          (float*)d_out, Nn);
}
